// Autoformer_30021821399877
// MI455X (gfx1250) — matrix operand order repaired and checked by reading, not compiled, not run on hardware
//
#include <hip/hip_runtime.h>
#include <math.h>

typedef __attribute__((ext_vector_type(16))) _Float16 v16h;
typedef __attribute__((ext_vector_type(8)))  float    v8f;

#define B_    8
#define LE_   1023
#define LD_   513
#define CM_   512
#define DFF_  2048
#define NV_   21
#define KTOP_E 20
#define KTOP_D 18

__device__ __forceinline__ float gelu_f(float x) {
  return 0.5f * x * (1.0f + tanhf(0.7978845608028654f * (x + 0.044715f * x * x * x)));
}

// ---------------------------------------------------------------------------
// WMMA GEMM:  C[M,N] = act(A[M,K] * W[N,K]^T + bias) + resid
// 128 threads = 4 waves; block tile 64x64; K-tile 64 staged in LDS as f16.
// Out-of-range tile rows are CLAMPED (not masked): addresses stay valid and
// divergence-free; the garbage rows only feed output elements that the store
// guard drops. 8 WMMAs per barrier pair.
// ---------------------------------------------------------------------------
__global__ __launch_bounds__(128)
void gemm_wmma_kernel(const float* __restrict__ A, const float* __restrict__ W,
                      const float* __restrict__ bias, const float* __restrict__ resid,
                      float* __restrict__ C, int M, int N, int K, int actGelu) {
  __shared__ _Float16 As[64][72];
  __shared__ _Float16 Ws[64][72];
  const int tid  = threadIdx.x;
  const int lane = tid & 31;
  const int wv   = tid >> 5;
  const int m0   = blockIdx.y << 6;
  const int n0   = blockIdx.x << 6;
  const int h    = lane >> 4;     // K-half selector within a 32-wide fragment
  const int lx   = lane & 15;

  v8f acc[4];
#pragma unroll
  for (int j = 0; j < 4; ++j)
#pragma unroll
    for (int i = 0; i < 8; ++i) acc[j][i] = 0.0f;

  for (int kb = 0; kb < K; kb += 64) {
    // cooperative stage: 64 rows x 64 cols f32 -> f16 for both A and W
#pragma unroll
    for (int i = 0; i < 8; ++i) {
      int lin = tid + (i << 7);            // 0..1023
      int r   = lin >> 4;                  // row 0..63
      int c4  = (lin & 15) << 2;           // col 0..60 step 4
      int gm  = m0 + r; if (gm >= M) gm = M - 1;      // clamp, no divergence
      float4 fa = *(const float4*)(A + (size_t)gm * K + kb + c4);
      As[r][c4 + 0] = (_Float16)fa.x; As[r][c4 + 1] = (_Float16)fa.y;
      As[r][c4 + 2] = (_Float16)fa.z; As[r][c4 + 3] = (_Float16)fa.w;
      int gn = n0 + r; if (gn >= N) gn = N - 1;
      float4 fw = *(const float4*)(W + (size_t)gn * K + kb + c4);
      Ws[r][c4 + 0] = (_Float16)fw.x; Ws[r][c4 + 1] = (_Float16)fw.y;
      Ws[r][c4 + 2] = (_Float16)fw.z; Ws[r][c4 + 3] = (_Float16)fw.w;
    }
    if (kb + 64 < K) {                     // uniform scalar branch: 1 per tile
#pragma unroll
      for (int i = 0; i < 8; i += 2) {
        int lin = tid + (i << 7);
        int r   = lin >> 4;
        int c4  = (lin & 15) << 2;
        int gm  = m0 + r; if (gm >= M) gm = M - 1;
        int gn  = n0 + r; if (gn >= N) gn = N - 1;
        __builtin_prefetch(A + (size_t)gm * K + kb + 64 + c4, 0, 1);
        __builtin_prefetch(W + (size_t)gn * K + kb + 64 + c4, 0, 1);
      }
    }
    __syncthreads();

#pragma unroll
    for (int ks = 0; ks < 64; ks += 32) {
      // A fragment (16x32): lane holds row (wv*16 + lx), K per ISA 7.12.2
      v16h af;
      const int ml = (wv << 4) + lx;
#pragma unroll
      for (int e = 0; e < 16; ++e) {
        int v  = e >> 1;
        int kk = ((v < 4) ? (v << 1) : (8 + (v << 1))) + (h << 3) + (e & 1);
        af[e] = As[ml][ks + kk];
      }
#pragma unroll
      for (int j = 0; j < 4; ++j) {
        v16h bf;
        const int nl = (j << 4) + lx;      // B[k][n] = W[n][k]
#pragma unroll
        for (int e = 0; e < 16; ++e) bf[e] = Ws[nl][ks + e + (h << 3) + ((e >> 3) << 3)];
        acc[j] = __builtin_amdgcn_wmma_f32_16x16x32_f16(
            false, af, false, bf, (short)0, acc[j], false, false);
      }
    }
    __syncthreads();
  }

  // store: D 16x16 f32 layout: VGPR d -> M = d + 8*h, N = lx
#pragma unroll
  for (int j = 0; j < 4; ++j) {
#pragma unroll
    for (int d = 0; d < 8; ++d) {
      int m = m0 + (wv << 4) + d + (h << 3);
      int n = n0 + (j << 4) + lx;
      if (m < M && n < N) {
        float val = acc[j][d];
        if (bias)    val += bias[n];
        if (actGelu) val = gelu_f(val);
        if (resid)   val += resid[(size_t)m * N + n];
        C[(size_t)m * N + n] = val;
      }
    }
  }
}

// ---------------------------------------------------------------------------
// mean_corr[b,tau] = (1/512) * sum_{t,c} q[b,t,c] * k[b,(t-tau)%L,c]
// ---------------------------------------------------------------------------
__global__ void meancorr_kernel(const float* __restrict__ q, const float* __restrict__ k,
                                float* __restrict__ out, int L, int Cc,
                                long qbs, long kbs, float scale) {
  const int tau = blockIdx.x, b = blockIdx.y;
  const float* qb = q + (long)b * qbs;
  const float* kb = k + (long)b * kbs;
  float acc = 0.0f;
  for (int t = threadIdx.x; t < L; t += blockDim.x) {
    int t2 = t - tau; if (t2 < 0) t2 += L;
    const float4* qr = (const float4*)(qb + (size_t)t * Cc);
    const float4* kr = (const float4*)(kb + (size_t)t2 * Cc);
    for (int c = 0; c < (Cc >> 2); ++c) {
      float4 a = qr[c], d = kr[c];
      acc += a.x * d.x + a.y * d.y + a.z * d.z + a.w * d.w;
    }
  }
  __shared__ float red[256];
  red[threadIdx.x] = acc; __syncthreads();
  for (int s = 128; s > 0; s >>= 1) {
    if (threadIdx.x < s) red[threadIdx.x] += red[threadIdx.x + s];
    __syncthreads();
  }
  if (threadIdx.x == 0) out[(long)b * L + tau] = red[0] * scale;
}

// ---------------------------------------------------------------------------
// Iterative top-K (desc, ties -> lower index) + softmax. One block per batch.
// ---------------------------------------------------------------------------
__global__ void topk_softmax_kernel(const float* __restrict__ mc, float* __restrict__ wout,
                                    int* __restrict__ dout, int L, int K) {
  const int b = blockIdx.x, t = threadIdx.x;
  __shared__ float vals[1024];
  __shared__ float rv[256];
  __shared__ int   ri[256];
  __shared__ float topv[32];
  __shared__ int   topi[32];
  for (int i = t; i < 1024; i += blockDim.x)
    vals[i] = (i < L) ? mc[(long)b * L + i] : -3.4e38f;
  __syncthreads();
  for (int kk = 0; kk < K; ++kk) {
    float best = -3.4e38f; int bi = 0;
    for (int i = t; i < L; i += blockDim.x)
      if (vals[i] > best) { best = vals[i]; bi = i; }
    rv[t] = best; ri[t] = bi; __syncthreads();
    for (int s = 128; s > 0; s >>= 1) {
      if (t < s) {
        if (rv[t + s] > rv[t] || (rv[t + s] == rv[t] && ri[t + s] < ri[t])) {
          rv[t] = rv[t + s]; ri[t] = ri[t + s];
        }
      }
      __syncthreads();
    }
    if (t == 0) { topv[kk] = rv[0]; topi[kk] = ri[0]; vals[ri[0]] = -3.4e38f; }
    __syncthreads();
  }
  if (t == 0) {
    float mx = topv[0], s = 0.0f;
    for (int i = 0; i < K; ++i) { float e = expf(topv[i] - mx); topv[i] = e; s += e; }
    for (int i = 0; i < K; ++i) { wout[b * 32 + i] = topv[i] / s; dout[b * 32 + i] = topi[i]; }
  }
}

// out[b,l,c] = sum_k w[b,k] * v[b, (l+delay_k)%L, c]
__global__ void aggregate_kernel(const float* __restrict__ v, const float* __restrict__ w,
                                 const int* __restrict__ d, float* __restrict__ out,
                                 int L, int Cc, long vbs, int K, long total) {
  long idx = (long)blockIdx.x * blockDim.x + threadIdx.x;
  if (idx >= total) return;
  int c = (int)(idx % Cc);
  long r = idx / Cc;
  int l = (int)(r % L);
  int b = (int)(r / L);
  float acc = 0.0f;
  for (int kk = 0; kk < K; ++kk) {
    int tt = l + d[b * 32 + kk]; if (tt >= L) tt -= L;
    acc += w[b * 32 + kk] * v[(long)b * vbs + (size_t)tt * Cc + c];
  }
  out[idx] = acc;
}

// ---------------------------------------------------------------------------
// series_decomp with 25-tap replicate-pad moving average; input may be x or x+y.
// tmode: 0 = no trend, 1 = trend store, 2 = trend accumulate
// ---------------------------------------------------------------------------
__global__ void decomp_kernel(const float* __restrict__ x, const float* __restrict__ y,
                              float* __restrict__ season, float* __restrict__ trend,
                              int L, int Cc, long xbs, int tmode, long total) {
  long idx = (long)blockIdx.x * blockDim.x + threadIdx.x;
  if (idx >= total) return;
  int c = (int)(idx % Cc);
  long r = idx / Cc;
  int l = (int)(r % L);
  int b = (int)(r / L);
  const float* xb = x + (long)b * xbs;
  const float* yb = y ? (y + (long)b * L * Cc) : nullptr;
  float s = 0.0f;
#pragma unroll
  for (int j = -12; j <= 12; ++j) {
    int ll = l + j; ll = ll < 0 ? 0 : (ll >= L ? L - 1 : ll);
    float vv = xb[(long)ll * Cc + c];
    if (yb) vv += yb[(long)ll * Cc + c];
    s += vv;
  }
  float tr  = s * (1.0f / 25.0f);
  float ctr = xb[(long)l * Cc + c] + (yb ? yb[(long)l * Cc + c] : 0.0f);
  season[idx] = ctr - tr;
  if (tmode == 1)      trend[idx]  = tr;
  else if (tmode == 2) trend[idx] += tr;
}

// Circular conv1d k=3 (+ optional positional embedding, + optional residual)
__global__ void token_conv_kernel(const float* __restrict__ x, const float* __restrict__ w,
                                  const float* __restrict__ resid, float* __restrict__ out,
                                  int L, int Cin, int Cout, long xbs, int addPE, long total) {
  long idx = (long)blockIdx.x * blockDim.x + threadIdx.x;
  if (idx >= total) return;
  int o = (int)(idx % Cout);
  long r = idx / Cout;
  int l = (int)(r % L);
  int b = (int)(r / L);
  int lm = (l == 0) ? (L - 1) : (l - 1);
  int lp = (l == L - 1) ? 0 : (l + 1);
  const float* x0 = x + (long)b * xbs;
  float acc = 0.0f;
  for (int c = 0; c < Cin; ++c) {
    const float* wr = w + ((size_t)o * Cin + c) * 3;
    acc += x0[(size_t)lm * Cin + c] * wr[0] +
           x0[(size_t)l  * Cin + c] * wr[1] +
           x0[(size_t)lp * Cin + c] * wr[2];
  }
  if (addPE) {
    int i2 = o & ~1;
    float div = expf(-(float)i2 * (9.210340371976184f / 512.0f)); // ln(1e4)/d
    float ang = (float)l * div;
    acc += (o & 1) ? cosf(ang) : sinf(ang);
  }
  if (resid) acc += resid[idx];
  out[idx] = acc;
}

// LayerNorm over channels; one block per row
__global__ void layernorm_kernel(const float* __restrict__ x, const float* __restrict__ g,
                                 const float* __restrict__ bta, float* __restrict__ out, int Cc) {
  long row = blockIdx.x;
  const float* xr = x + row * (long)Cc;
  __shared__ float s1[256], s2[256];
  float a = 0.0f, q = 0.0f;
  for (int c = threadIdx.x; c < Cc; c += 256) { float v = xr[c]; a += v; q += v * v; }
  s1[threadIdx.x] = a; s2[threadIdx.x] = q; __syncthreads();
  for (int s = 128; s > 0; s >>= 1) {
    if (threadIdx.x < s) { s1[threadIdx.x] += s1[threadIdx.x + s]; s2[threadIdx.x] += s2[threadIdx.x + s]; }
    __syncthreads();
  }
  float mu = s1[0] / Cc;
  float var = s2[0] / Cc - mu * mu;
  float rr = rsqrtf(var + 1e-5f);
  for (int c = threadIdx.x; c < Cc; c += 256)
    out[row * (long)Cc + c] = (xr[c] - mu) * rr * g[c] + bta[c];
}

// subtract per-(b,c) mean over L (second half of my_layernorm); in-place
__global__ void submean_l_kernel(float* __restrict__ x, int L, int Cc) {
  int b = blockIdx.x / Cc, c = blockIdx.x % Cc;
  float* base = x + (long)b * L * Cc + c;
  __shared__ float s1[256];
  float a = 0.0f;
  for (int l = threadIdx.x; l < L; l += 256) a += base[(size_t)l * Cc];
  s1[threadIdx.x] = a; __syncthreads();
  for (int s = 128; s > 0; s >>= 1) {
    if (threadIdx.x < s) s1[threadIdx.x] += s1[threadIdx.x + s];
    __syncthreads();
  }
  float mu = s1[0] / L;
  for (int l = threadIdx.x; l < L; l += 256) base[(size_t)l * Cc] -= mu;
}

// per-(b,c) mean over L rows
__global__ void colmean_kernel(const float* __restrict__ x, float* __restrict__ out,
                               int L, int Cc, long xbs) {
  int b = blockIdx.x / Cc, c = blockIdx.x % Cc;
  const float* base = x + (long)b * xbs + c;
  __shared__ float s1[256];
  float a = 0.0f;
  for (int l = threadIdx.x; l < L; l += 256) a += base[(size_t)l * Cc];
  s1[threadIdx.x] = a; __syncthreads();
  for (int s = 128; s > 0; s >>= 1) {
    if (threadIdx.x < s) s1[threadIdx.x] += s1[threadIdx.x + s];
    __syncthreads();
  }
  if (threadIdx.x == 0) out[blockIdx.x] = s1[0] / L;
}

// per-(b,c) standardization over the full 1024 length (+ finite guard)
__global__ void norm_input_kernel(const float* __restrict__ xin, float* __restrict__ xb) {
  int b = blockIdx.x / NV_, c = blockIdx.x % NV_;
  const float* base = xin + (long)b * 1024 * NV_ + c;
  float* ob = xb + (long)b * 1024 * NV_ + c;
  __shared__ float s1[256], s2[256];
  float a = 0.0f, q = 0.0f;
  for (int l = threadIdx.x; l < 1024; l += 256) {
    float v = base[(size_t)l * NV_]; a += v; q += v * v;
  }
  s1[threadIdx.x] = a; s2[threadIdx.x] = q; __syncthreads();
  for (int s = 128; s > 0; s >>= 1) {
    if (threadIdx.x < s) { s1[threadIdx.x] += s1[threadIdx.x + s]; s2[threadIdx.x] += s2[threadIdx.x + s]; }
    __syncthreads();
  }
  float mu = s1[0] / 1024.0f;
  float var = s2[0] / 1024.0f - mu * mu;
  float rr = rsqrtf(var);
  for (int l = threadIdx.x; l < 1024; l += 256) {
    float v = (base[(size_t)l * NV_] - mu) * rr;
    if (!(v > -3.0e38f && v < 3.0e38f)) v = 0.0f;   // where(isfinite, v, 0)
    ob[(size_t)l * NV_] = v;
  }
}

// seasonal_init / trend_init construction
__global__ void build_init_kernel(const float* __restrict__ sf, const float* __restrict__ tf,
                                  const float* __restrict__ xm,
                                  float* __restrict__ sinit, float* __restrict__ tinit) {
  long idx = (long)blockIdx.x * blockDim.x + threadIdx.x;
  long total = (long)B_ * LD_ * NV_;
  if (idx >= total) return;
  int c = (int)(idx % NV_);
  long r = idx / NV_;
  int l = (int)(r % LD_);
  int b = (int)(r / LD_);
  if (l < LD_ - 1) {
    long src = ((long)b * LE_ + (l + 511)) * NV_ + c;   // last 512 of 1023
    sinit[idx] = sf[src];
    tinit[idx] = tf[src];
  } else {
    sinit[idx] = 0.0f;
    tinit[idx] = xm[b * NV_ + c];
  }
}

// d_out = [output[:, -1, :] (168) | gt (168)]
__global__ void final_kernel(const float* __restrict__ tr, const float* __restrict__ sp,
                             const float* __restrict__ xb, float* __restrict__ out) {
  int idx = blockIdx.x * blockDim.x + threadIdx.x;
  if (idx < B_ * NV_) {
    int b = idx / NV_, c = idx % NV_;
    long p = ((long)b * LD_ + (LD_ - 1)) * NV_ + c;
    out[idx] = tr[p] + sp[p];
  } else if (idx < 2 * B_ * NV_) {
    int j = idx - B_ * NV_;
    int b = j / NV_, c = j % NV_;
    out[idx] = xb[((long)b * 1024 + 1023) * NV_ + c];
  }
}

// ===========================================================================
extern "C" void kernel_launch(void* const* d_in, const int* in_sizes, int n_in,
                              void* d_out, int out_size, void* d_ws, size_t ws_size,
                              hipStream_t stream) {
  (void)in_sizes; (void)n_in; (void)out_size; (void)ws_size;

  // ---- inputs (setup_inputs dict order, depth-first) ----
  int p = 0;
  const float* x_enc = (const float*)d_in[p++];
  p++; // drop scalar
  const float* enc_emb = (const float*)d_in[p++];
  const float* dec_emb = (const float*)d_in[p++];
  const float* enc_nw  = (const float*)d_in[p++];
  const float* enc_nb  = (const float*)d_in[p++];
  const float* dec_nw  = (const float*)d_in[p++];
  const float* dec_nb  = (const float*)d_in[p++];
  const float* proj_w  = (const float*)d_in[p++];
  const float* proj_b  = (const float*)d_in[p++];
  const float* EL[2][10];
  for (int l = 0; l < 2; ++l)
    for (int i = 0; i < 10; ++i) EL[l][i] = (const float*)d_in[p++];
  const float* DL[19];
  for (int i = 0; i < 19; ++i) DL[i] = (const float*)d_in[p++];
  // EL: 0 wq 1 bq 2 wk 3 bk 4 wv 5 bv 6 wo 7 bo 8 conv1 9 conv2
  // DL: 0..7 sa_{wq,bq,wk,bk,wv,bv,wo,bo}; 8..15 ca_*; 16 conv1; 17 conv2; 18 trend_proj

  // ---- workspace bump allocator ----
  char* base = (char*)d_ws;
  size_t off = 0;
  auto alloc = [&](size_t nfloats) -> float* {
    float* r = (float*)(base + off);
    off += (nfloats * sizeof(float) + 255) & ~(size_t)255;
    return r;
  };
  float* XB    = alloc((size_t)B_ * 1024 * NV_);
  float* SFULL = alloc((size_t)B_ * LE_ * NV_);
  float* TFULL = alloc((size_t)B_ * LE_ * NV_);
  float* XMEAN = alloc((size_t)B_ * NV_);
  float* SINIT = alloc((size_t)B_ * LD_ * NV_);
  float* TINIT = alloc((size_t)B_ * LD_ * NV_);
  float* ENC   = alloc((size_t)B_ * LE_ * CM_);
  float* CROSS = alloc((size_t)B_ * LE_ * CM_);
  float* BQ    = alloc((size_t)B_ * LE_ * CM_);
  float* BK    = alloc((size_t)B_ * LE_ * CM_);
  float* BV    = alloc((size_t)B_ * LE_ * CM_);
  float* AGG   = alloc((size_t)B_ * LE_ * CM_);
  float* TBUF  = alloc((size_t)B_ * LE_ * CM_);
  float* FF    = alloc((size_t)B_ * LE_ * DFF_);
  float* DEC   = alloc((size_t)B_ * LD_ * CM_);
  float* DEC2  = alloc((size_t)B_ * LD_ * CM_);
  float* TSUM  = alloc((size_t)B_ * LD_ * CM_);
  float* TREND = alloc((size_t)B_ * LD_ * NV_);
  float* SPART = alloc((size_t)B_ * LD_ * NV_);
  float* MC    = alloc((size_t)B_ * LE_);
  float* WTOP  = alloc(B_ * 32);
  int*   DTOP  = (int*)alloc(B_ * 32);

  auto gemm = [&](const float* A, const float* W, const float* bias, const float* resid,
                  float* O, int M, int N, int K, int act) {
    dim3 g((N + 63) / 64, (M + 63) / 64);
    gemm_wmma_kernel<<<g, 128, 0, stream>>>(A, W, bias, resid, O, M, N, K, act);
  };
  auto decomp = [&](const float* x, const float* y, float* season, float* trend,
                    int L, int Cc, long xbs, int tmode) {
    long total = (long)B_ * L * Cc;
    decomp_kernel<<<(unsigned)((total + 255) / 256), 256, 0, stream>>>(
        x, y, season, trend, L, Cc, xbs, tmode, total);
  };
  auto tconv = [&](const float* x, const float* w, const float* resid, float* o,
                   int L, int Cin, int Cout, long xbs, int addPE) {
    long total = (long)B_ * L * Cout;
    token_conv_kernel<<<(unsigned)((total + 255) / 256), 256, 0, stream>>>(
        x, w, resid, o, L, Cin, Cout, xbs, addPE, total);
  };
  auto attention = [&](const float* qin, int Lq, const float* kvin, int Lkv,
                       const float* wq, const float* bq, const float* wk, const float* bk,
                       const float* wv, const float* bv, int Ktop, float* outAgg) {
    int Mq = B_ * Lq, Mkv = B_ * Lkv;
    gemm(qin,  wq, bq, nullptr, BQ, Mq,  CM_, CM_, 0);
    gemm(kvin, wk, bk, nullptr, BK, Mkv, CM_, CM_, 0);
    gemm(kvin, wv, bv, nullptr, BV, Mkv, CM_, CM_, 0);
    int L = Lq; // L <= Lkv in all uses (k,v truncated to L)
    meancorr_kernel<<<dim3((unsigned)L, B_), 256, 0, stream>>>(
        BQ, BK, MC, L, CM_, (long)Lq * CM_, (long)Lkv * CM_, 1.0f / 512.0f);
    topk_softmax_kernel<<<B_, 256, 0, stream>>>(MC, WTOP, DTOP, L, Ktop);
    long total = (long)B_ * L * CM_;
    aggregate_kernel<<<(unsigned)((total + 255) / 256), 256, 0, stream>>>(
        BV, WTOP, DTOP, outAgg, L, CM_, (long)Lkv * CM_, Ktop, total);
  };

  // ---- 0) input standardization + finite guard ----
  norm_input_kernel<<<B_ * NV_, 256, 0, stream>>>(x_enc, XB);

  // ---- 1) initial decomposition of x_input (rows 0..1022 of XB) ----
  decomp(XB, nullptr, SFULL, TFULL, LE_, NV_, (long)1024 * NV_, 1);
  colmean_kernel<<<B_ * NV_, 256, 0, stream>>>(XB, XMEAN, LE_, NV_, (long)1024 * NV_);
  {
    long total = (long)B_ * LD_ * NV_;
    build_init_kernel<<<(unsigned)((total + 255) / 256), 256, 0, stream>>>(
        SFULL, TFULL, XMEAN, SINIT, TINIT);
  }

  // ---- 2) encoder embedding + encoder layers ----
  tconv(XB, enc_emb, nullptr, ENC, LE_, NV_, CM_, (long)1024 * NV_, 0);
  const int ME = B_ * LE_;
  for (int l = 0; l < 2; ++l) {
    attention(ENC, LE_, ENC, LE_, EL[l][0], EL[l][1], EL[l][2], EL[l][3],
              EL[l][4], EL[l][5], KTOP_E, AGG);
    gemm(AGG, EL[l][6], EL[l][7], ENC, TBUF, ME, CM_, CM_, 0);      // x + attn
    decomp(TBUF, nullptr, ENC, nullptr, LE_, CM_, (long)LE_ * CM_, 0);
    gemm(ENC, EL[l][8], nullptr, nullptr, FF, ME, DFF_, CM_, 1);    // gelu(x@c1^T)
    gemm(FF, EL[l][9], nullptr, ENC, TBUF, ME, CM_, DFF_, 0);       // x + y
    decomp(TBUF, nullptr, ENC, nullptr, LE_, CM_, (long)LE_ * CM_, 0);
  }
  layernorm_kernel<<<ME, 256, 0, stream>>>(ENC, enc_nw, enc_nb, CROSS, CM_);
  submean_l_kernel<<<B_ * CM_, 256, 0, stream>>>(CROSS, LE_, CM_);

  // ---- 3) decoder embedding (+ positional embedding) ----
  tconv(SINIT, dec_emb, nullptr, DEC, LD_, NV_, CM_, (long)LD_ * NV_, 1);

  // ---- 4) decoder layer ----
  const int MD = B_ * LD_;
  attention(DEC, LD_, DEC, LD_, DL[0], DL[1], DL[2], DL[3], DL[4], DL[5], KTOP_D, AGG);
  gemm(AGG, DL[6], DL[7], DEC, DEC2, MD, CM_, CM_, 0);
  decomp(DEC2, nullptr, DEC, TSUM, LD_, CM_, (long)LD_ * CM_, 1);   // t1
  attention(DEC, LD_, CROSS, LE_, DL[8], DL[9], DL[10], DL[11], DL[12], DL[13], KTOP_D, AGG);
  gemm(AGG, DL[14], DL[15], DEC, DEC2, MD, CM_, CM_, 0);
  decomp(DEC2, nullptr, DEC, TSUM, LD_, CM_, (long)LD_ * CM_, 2);   // +t2
  gemm(DEC, DL[16], nullptr, nullptr, FF, MD, DFF_, CM_, 1);
  gemm(FF, DL[17], nullptr, DEC, DEC2, MD, CM_, DFF_, 0);
  decomp(DEC2, nullptr, DEC, TSUM, LD_, CM_, (long)LD_ * CM_, 2);   // +t3
  tconv(TSUM, DL[18], TINIT, TREND, LD_, CM_, NV_, (long)LD_ * CM_, 0); // trend_init + token_conv(t1+t2+t3)

  // ---- 5) decoder norm + projection + output ----
  layernorm_kernel<<<MD, 256, 0, stream>>>(DEC, dec_nw, dec_nb, DEC2, CM_);
  submean_l_kernel<<<B_ * CM_, 256, 0, stream>>>(DEC2, LD_, CM_);
  gemm(DEC2, proj_w, proj_b, nullptr, SPART, MD, NV_, CM_, 0);
  final_kernel<<<2, 256, 0, stream>>>(TREND, SPART, XB, (float*)d_out);
}
